// FastChannelAttention_76295799046376
// MI455X (gfx1250) — matrix operand order repaired and checked by reading, not compiled, not run on hardware
//
#include <hip/hip_runtime.h>
#include <hip/hip_bf16.h>
#include <math.h>

// ---------- types ----------
typedef __attribute__((ext_vector_type(16))) __bf16        bf16x16;
typedef __attribute__((ext_vector_type(8)))  float         v8f;
typedef __attribute__((ext_vector_type(4)))  unsigned int  u32x4;

union Frag { u32x4 u[2]; bf16x16 v; };

// ---------- helpers ----------
__device__ __forceinline__ unsigned short f2bf(float f) {
    unsigned u = __float_as_uint(f);
    u += 0x7FFFu + ((u >> 16) & 1u);      // round-to-nearest-even
    return (unsigned short)(u >> 16);
}
__device__ __forceinline__ unsigned pk2(float a, float b) {
    return (unsigned)f2bf(a) | ((unsigned)f2bf(b) << 16);
}
__device__ __forceinline__ v8f wmma_bf16(const Frag& a, const Frag& b, v8f c) {
    return __builtin_amdgcn_wmma_f32_16x16x32_bf16(false, a.v, false, b.v,
                                                   (short)0, c, false, false);
}

// ---------- 1a: x (f32) -> bf16 ----------
__global__ __launch_bounds__(256)
void convert_x(const float* __restrict__ x, unsigned short* __restrict__ xbf, int n8) {
    int i = blockIdx.x * 256 + threadIdx.x;
    if (i >= n8) return;
    const float* p = x + (size_t)i * 8;
    u32x4 pk = { pk2(p[0], p[1]), pk2(p[2], p[3]), pk2(p[4], p[5]), pk2(p[6], p[7]) };
    *(u32x4*)(xbf + (size_t)i * 8) = pk;
}

// ---------- 1b: weight [512][Ccols] f32 -> wT [Ccols][512] bf16 ----------
__global__ __launch_bounds__(256)
void transpose_w(const float* __restrict__ w, unsigned short* __restrict__ wT,
                 int Ccols, int total) {
    int e = blockIdx.x * 256 + threadIdx.x;
    if (e >= total) return;
    int c = e >> 9;            // K = 512
    int k = e & 511;
    wT[e] = f2bf(w[(size_t)k * Ccols + c]);
}

// ---------- 2/6: double-buffered LDS-tiled bf16 WMMA GEMM, K=512, 128x128 tile ----------
// MODE 0: qkv  -> q,k transposed bf16 [b*512+ch][16384], v token-major bf16
// MODE 1: proj -> f32 out + bias, token-major
template<int MODE>
__global__ __launch_bounds__(256)
void gemm512(const unsigned short* __restrict__ A,   // [Mtok][512] bf16
             const unsigned short* __restrict__ BT,  // [Ncols][512] bf16
             unsigned short* __restrict__ qT,
             unsigned short* __restrict__ kT,
             unsigned short* __restrict__ vtok,
             const float* __restrict__ bias,
             float* __restrict__ Y)
{
    // padded row stride 40 (80B = 5*16B) -> conflict-free ds_load_b128
    __shared__ unsigned short As[2][128 * 40];
    __shared__ unsigned short Bs[2][128 * 40];
    const int tid  = threadIdx.x;
    const int mb   = blockIdx.x * 128;        // token base
    const int nb   = blockIdx.y * 128;        // output-channel base
    const int wave = tid >> 5, lane = tid & 31;
    const int wm   = (wave >> 1) * 32;
    const int wn   = (wave & 1) * 64;
    const int l15  = lane & 15;
    const int alo  = (lane < 16) ? 0 : 8;
    const int blo  = (lane < 16) ? 0 : 8;

    // staging: each thread owns 2 16B chunks of A and 2 of B per tile
    const int c0  = tid,      r0 = c0 >> 2,  k0 = (c0 & 3) << 3;
    const int c1  = tid + 256,r1 = c1 >> 2,  k1 = (c1 & 3) << 3;

    u32x4 ra0, ra1, rb0, rb1;
    auto glbload = [&](int kk) {
        ra0 = *(const u32x4*)(A  + (size_t)(mb + r0) * 512 + kk + k0);
        ra1 = *(const u32x4*)(A  + (size_t)(mb + r1) * 512 + kk + k1);
        rb0 = *(const u32x4*)(BT + (size_t)(nb + r0) * 512 + kk + k0);
        rb1 = *(const u32x4*)(BT + (size_t)(nb + r1) * 512 + kk + k1);
        if (kk + 32 < 512) {  // gfx1250 global_prefetch_b8, one tile further ahead
            __builtin_prefetch(A  + (size_t)(mb + r0) * 512 + kk + 32 + k0, 0, 0);
            __builtin_prefetch(BT + (size_t)(nb + r0) * 512 + kk + 32 + k0, 0, 0);
        }
    };
    auto dsstore = [&](int buf) {
        *(u32x4*)(As[buf] + r0 * 40 + k0) = ra0;
        *(u32x4*)(As[buf] + r1 * 40 + k1) = ra1;
        *(u32x4*)(Bs[buf] + r0 * 40 + k0) = rb0;
        *(u32x4*)(Bs[buf] + r1 * 40 + k1) = rb1;
    };

    v8f acc[2][4];
    #pragma unroll
    for (int i = 0; i < 2; ++i)
        #pragma unroll
        for (int j = 0; j < 4; ++j) acc[i][j] = (v8f)0.0f;

    glbload(0);
    dsstore(0);
    #pragma unroll 1
    for (int t = 0; t < 16; ++t) {
        const int cur = t & 1;
        if (t < 15) glbload((t + 1) * 32);     // in flight during this tile's WMMAs
        __syncthreads();                       // buf[cur] stores visible, prev reads done
        Frag a[2];
        #pragma unroll
        for (int mi = 0; mi < 2; ++mi) {
            int r = wm + mi * 16 + l15;
            a[mi].u[0] = *(const u32x4*)(As[cur] + r * 40 + alo);
            a[mi].u[1] = *(const u32x4*)(As[cur] + r * 40 + alo + 16);
        }
        #pragma unroll
        for (int ni = 0; ni < 4; ++ni) {
            int r = wn + ni * 16 + l15;
            Frag b;
            b.u[0] = *(const u32x4*)(Bs[cur] + r * 40 + blo);
            b.u[1] = *(const u32x4*)(Bs[cur] + r * 40 + blo + 16);
            #pragma unroll
            for (int mi = 0; mi < 2; ++mi)
                acc[mi][ni] = wmma_bf16(a[mi], b, acc[mi][ni]);
        }
        if (t < 15) dsstore(cur ^ 1);          // safe: last read of buf[cur^1] was before barrier
    }

    // Epilogue: lane holds 8 consecutive token-rows of one column.
    #pragma unroll
    for (int mi = 0; mi < 2; ++mi) {
        int t0 = mb + wm + mi * 16 + ((lane >> 4) << 3);
        #pragma unroll
        for (int ni = 0; ni < 4; ++ni) {
            v8f v = acc[mi][ni];
            if (MODE == 0) {
                int seg = nb >> 9;                          // 0=q 1=k 2=v
                int c   = (nb & 511) + wn + ni * 16 + l15;
                if (seg < 2) {
                    int b = t0 >> 14, n0 = t0 & 16383;
                    unsigned short* dst = (seg == 0 ? qT : kT)
                        + (((size_t)((b << 9) + c)) << 14) + n0;
                    u32x4 pk = { pk2(v[0], v[1]), pk2(v[2], v[3]),
                                 pk2(v[4], v[5]), pk2(v[6], v[7]) };
                    *(u32x4*)dst = pk;                      // 16B transposed store
                } else {
                    #pragma unroll
                    for (int r = 0; r < 8; ++r)
                        vtok[(size_t)(t0 + r) * 512 + c] = f2bf(v[r]);
                }
            } else {
                int c = nb + wn + ni * 16 + l15;
                float bb = bias[c];
                #pragma unroll
                for (int r = 0; r < 8; ++r)
                    Y[(size_t)(t0 + r) * 512 + c] = v[r] + bb;
            }
        }
    }
}

// ---------- 3: per-channel L2 norms over N=16384 tokens ----------
__global__ __launch_bounds__(256)
void colnorms(const unsigned short* __restrict__ qT, const unsigned short* __restrict__ kT,
              float* __restrict__ nq, float* __restrict__ nk) {
    int row = blockIdx.x;    // 0..4095 : 0..2047 q rows, 2048..4095 k rows
    const unsigned short* src = ((row < 2048) ? qT : kT) + (size_t)(row & 2047) * 16384;
    int tid = threadIdx.x;
    float s = 0.f;
    for (int it = 0; it < 8; ++it) {
        int base = (tid + it * 256) * 8;
        u32x4 d = *(const u32x4*)(src + base);
        #pragma unroll
        for (int j = 0; j < 4; ++j) {
            unsigned u = d[j];
            float lo = __uint_as_float(u << 16);
            float hi = __uint_as_float(u & 0xFFFF0000u);
            s += lo * lo + hi * hi;
        }
    }
    __shared__ float red[256];
    red[tid] = s; __syncthreads();
    for (int o = 128; o > 0; o >>= 1) {
        if (tid < o) red[tid] += red[tid + o];
        __syncthreads();
    }
    if (tid == 0) ((row < 2048) ? nq : nk)[row & 2047] = sqrtf(red[0]);
}

// ---------- 4a: split-K logits partials: Sp[split][bh][64][64] (f32) ----------
// grid (32 bh, 16 splits), 128 threads. Software-pipelined fragment loads.
__global__ __launch_bounds__(128)
void attn_partial(const unsigned short* __restrict__ qT, const unsigned short* __restrict__ kT,
                  float* __restrict__ Sp)
{
    const int bh = blockIdx.x, b = bh >> 3, h = bh & 7;
    const int split = blockIdx.y;
    const int tid = threadIdx.x, wave = tid >> 5, lane = tid & 31;
    const int l15 = lane & 15;
    const int alo = (lane < 16) ? 0 : 8;
    const int blo = (lane < 16) ? 0 : 8;
    const size_t rowbase = (size_t)(b * 512 + h * 64) * 16384;
    const unsigned short* qrow = qT + rowbase + (size_t)(wave * 16 + l15) * 16384;
    const unsigned short* kb   = kT + rowbase + (size_t)l15 * 16384;
    const int n0  = split * 1024;
    const int end = n0 + 1024;

    auto loadA = [&](Frag& a, int n) {
        a.u[0] = *(const u32x4*)(qrow + n + alo);
        a.u[1] = *(const u32x4*)(qrow + n + alo + 16);
    };
    auto loadB4 = [&](Frag (&bf)[4], int n) {
        #pragma unroll
        for (int ni = 0; ni < 4; ++ni) {
            const unsigned short* kr = kb + (size_t)(ni * 16) * 16384;
            bf[ni].u[0] = *(const u32x4*)(kr + n + blo);
            bf[ni].u[1] = *(const u32x4*)(kr + n + blo + 16);
        }
    };

    v8f acc[4];
    #pragma unroll
    for (int ni = 0; ni < 4; ++ni) acc[ni] = (v8f)0.0f;

    Frag a0, a1, b0[4], b1[4];
    loadA(a0, n0); loadB4(b0, n0);
    #pragma unroll 1
    for (int n = n0; n < end; n += 64) {       // 2 K-steps per trip, ping-pong registers
        loadA(a1, n + 32); loadB4(b1, n + 32); // issue before consuming a0/b0
        #pragma unroll
        for (int ni = 0; ni < 4; ++ni) acc[ni] = wmma_bf16(a0, b0[ni], acc[ni]);
        if (n + 64 < end) { loadA(a0, n + 64); loadB4(b0, n + 64); }
        #pragma unroll
        for (int ni = 0; ni < 4; ++ni) acc[ni] = wmma_bf16(a1, b1[ni], acc[ni]);
    }

    float* out = Sp + ((size_t)(split * 32 + bh)) * 4096;
    const int r0 = wave * 16 + ((lane >> 4) << 3);
    #pragma unroll
    for (int ni = 0; ni < 4; ++ni)
        #pragma unroll
        for (int r = 0; r < 8; ++r)
            out[(r0 + r) * 64 + ni * 16 + l15] = acc[ni][r];
}

// ---------- 4b: sum partials, scale by norms/temperature, softmax, bf16 attn ----------
__global__ __launch_bounds__(256)
void attn_finish(const float* __restrict__ Sp,
                 const float* __restrict__ nq, const float* __restrict__ nk,
                 const float* __restrict__ temp, unsigned short* __restrict__ attn)
{
    const int bh = blockIdx.x, b = bh >> 3, h = bh & 7;
    const int tid = threadIdx.x;
    __shared__ float S[64 * 65];
    __shared__ float invq[64], invk[64];
    for (int e = tid; e < 4096; e += 256) {
        float s = 0.f;
        #pragma unroll
        for (int sp = 0; sp < 16; ++sp)
            s += Sp[((size_t)(sp * 32 + bh)) * 4096 + e];
        S[(e >> 6) * 65 + (e & 63)] = s;
    }
    if (tid < 64) {
        int c = b * 512 + h * 64 + tid;
        invq[tid] = 1.f / fmaxf(nq[c], 1e-12f);
        invk[tid] = 1.f / fmaxf(nk[c], 1e-12f);
    }
    __syncthreads();
    if (tid < 64) {
        float* Sr = S + tid * 65;
        const float sq = invq[tid] * temp[h];
        float mx = -3.4e38f;
        for (int j = 0; j < 64; ++j) { float s = Sr[j] * sq * invk[j]; Sr[j] = s; mx = fmaxf(mx, s); }
        float sum = 0.f;
        for (int j = 0; j < 64; ++j) { float e = __expf(Sr[j] - mx); Sr[j] = e; sum += e; }
        float inv = 1.f / sum;
        unsigned short* dst = attn + (size_t)bh * 4096 + tid * 64;
        for (int j = 0; j < 64; ++j) dst[j] = f2bf(Sr[j] * inv);
    }
}

// ---------- 5: out[tok, d] = v[tok, e] @ attn^T (per head), K=64 ----------
__global__ __launch_bounds__(256)
void attn_v(const unsigned short* __restrict__ vtok, const unsigned short* __restrict__ attn,
            unsigned short* __restrict__ outtok)
{
    const int tb = blockIdx.x * 128;
    const int h  = blockIdx.y;
    const int tid = threadIdx.x, wave = tid >> 5, lane = tid & 31;
    const int l15 = lane & 15;
    const int alo = (lane < 16) ? 0 : 8;
    const int blo = (lane < 16) ? 0 : 8;
    const int b = tb >> 14;
    const unsigned short* vrow = vtok + (size_t)(tb + wave * 16 + l15) * 512 + h * 64;
    const unsigned short* Abh  = attn + (size_t)((b << 3) + h) * 4096;

    // Load ALL fragments up front (K=64 => 2 A + 8 B), then 8 back-to-back WMMAs.
    Frag a[2], bf[2][4];
    #pragma unroll
    for (int k = 0; k < 2; ++k) {
        const int kk = k * 32;
        a[k].u[0] = *(const u32x4*)(vrow + kk + alo);
        a[k].u[1] = *(const u32x4*)(vrow + kk + alo + 16);
        #pragma unroll
        for (int ni = 0; ni < 4; ++ni) {
            const unsigned short* br = Abh + (size_t)(ni * 16 + l15) * 64 + kk;
            bf[k][ni].u[0] = *(const u32x4*)(br + blo);
            bf[k][ni].u[1] = *(const u32x4*)(br + blo + 16);
        }
    }
    v8f acc[4];
    #pragma unroll
    for (int ni = 0; ni < 4; ++ni) acc[ni] = (v8f)0.0f;
    #pragma unroll
    for (int k = 0; k < 2; ++k)
        #pragma unroll
        for (int ni = 0; ni < 4; ++ni)
            acc[ni] = wmma_bf16(a[k], bf[k][ni], acc[ni]);

    const int t0 = tb + wave * 16 + ((lane >> 4) << 3);
    #pragma unroll
    for (int ni = 0; ni < 4; ++ni) {
        int c = h * 64 + ni * 16 + l15;
        #pragma unroll
        for (int r = 0; r < 8; ++r)
            outtok[(size_t)(t0 + r) * 512 + c] = f2bf(acc[ni][r]);
    }
}

// ---------- launch ----------
extern "C" void kernel_launch(void* const* d_in, const int* in_sizes, int n_in,
                              void* d_out, int out_size, void* d_ws, size_t ws_size,
                              hipStream_t stream)
{
    (void)in_sizes; (void)n_in; (void)out_size; (void)ws_size;
    // inputs: 0:x 1:H 2:W 3:qkv_w 4:proj_w 5:proj_b 6:temperature
    const float* x      = (const float*)d_in[0];
    const float* qkv_w  = (const float*)d_in[3];
    const float* proj_w = (const float*)d_in[4];
    const float* proj_b = (const float*)d_in[5];
    const float* temp   = (const float*)d_in[6];
    float* y = (float*)d_out;

    char* ws = (char*)d_ws;
    const size_t MB = 1ull << 20;
    unsigned short* xbf    = (unsigned short*)(ws + 0);         // 64MB (reused as out_tok)
    unsigned short* qT     = (unsigned short*)(ws + 64  * MB);  // 64MB  [b*512+ch][16384]
    unsigned short* kT     = (unsigned short*)(ws + 128 * MB);  // 64MB
    unsigned short* vtok   = (unsigned short*)(ws + 192 * MB);  // 64MB  [tok][512]
    unsigned short* qkvwT  = (unsigned short*)(ws + 256 * MB);  // 1.5MB [1536][512]
    unsigned short* projwT = (unsigned short*)(ws + 258 * MB);  // 0.5MB [512][512]
    float*          nq     = (float*)         (ws + 259 * MB);  // 8KB
    float*          nk     = (float*)         (ws + 259 * MB + 8192);
    unsigned short* attn   = (unsigned short*)(ws + 260 * MB);  // 256KB [32][64][64] bf16
    float*          Sp     = (float*)         (ws + 262 * MB);  // 8MB   [16][32][64][64] f32
    unsigned short* outtok = xbf;                               // alias: x_bf16 dead after QKV GEMM

    convert_x   <<<16384,          256, 0, stream>>>(x, xbf, 4194304);
    transpose_w <<<(512*1536)/256, 256, 0, stream>>>(qkv_w,  qkvwT,  1536, 512 * 1536);
    transpose_w <<<(512*512)/256,  256, 0, stream>>>(proj_w, projwT, 512,  512 * 512);
    gemm512<0>  <<<dim3(512, 12),  256, 0, stream>>>(xbf, qkvwT, qT, kT, vtok, nullptr, nullptr);
    colnorms    <<<4096,           256, 0, stream>>>(qT, kT, nq, nk);
    attn_partial<<<dim3(32, 16),   128, 0, stream>>>(qT, kT, Sp);
    attn_finish <<<32,             256, 0, stream>>>(Sp, nq, nk, temp, attn);
    attn_v      <<<dim3(512, 8),   256, 0, stream>>>(vtok, attn, outtok);
    gemm512<1>  <<<dim3(512, 4),   256, 0, stream>>>(outtok, projwT, nullptr, nullptr, nullptr, proj_b, y);
}
